// MultiHeadSelfAttention_48275432407814
// MI455X (gfx1250) — matrix operand order repaired and checked by reading, not compiled, not run on hardware
//
#include <hip/hip_runtime.h>
#include <hip/hip_bf16.h>
#include <cstdint>

// ---- problem constants (from reference setup_inputs) ----
#define B_   2
#define S_   2048
#define D_   1024
#define H_   16
#define QK_  64
#define BS_  (B_ * S_)      // 4096
#define N3_  (3 * D_)       // 3072 stacked QKV output columns

typedef __attribute__((ext_vector_type(16))) _Float16 v16h;
typedef __attribute__((ext_vector_type(8)))  _Float16 v8h;
typedef __attribute__((ext_vector_type(8)))  float    v8f;
typedef __attribute__((ext_vector_type(4)))  unsigned int v4u;
typedef __attribute__((ext_vector_type(8)))  int      v8i;
typedef __attribute__((ext_vector_type(4)))  int      v4i;

#if defined(__has_builtin)
#if __has_builtin(__builtin_amdgcn_tensor_load_to_lds) && __has_builtin(__builtin_amdgcn_s_wait_tensorcnt)
#define HAVE_TDM 1
#endif
#endif

// LDS row stride for staged B tiles: 32 halfs payload + 8 halfs pad (4 dwords)
#define BROW_ 40
#define BBUF_ (64 * BROW_)   // one buffer: 64 rows

// ---------------------------------------------------------------------------
// WMMA helper (wave32, 16x16x32 f16 -> f32 accumulate)
// ---------------------------------------------------------------------------
__device__ __forceinline__ v8f wmma16(v16h a, v16h b, v8f c) {
  return __builtin_amdgcn_wmma_f32_16x16x32_f16(
      false, a, false, b, (short)0, c, false, false);
}

// A-fragment: 16x32 (MxK) row-major, leading dim ld (halfs).
// lane m=L&15, half=L>>4; elems 0..15 = K(half*16..half*16+15)
__device__ __forceinline__ v16h load_a_frag(const _Float16* __restrict__ base,
                                            int ld, int lane) {
  const int m = lane & 15, hl = lane >> 4;
  const _Float16* p = base + (size_t)m * ld + hl * 16;
  union { v16h v; v8h h[2]; } u;
  u.h[0] = *(const v8h*)(p);
  u.h[1] = *(const v8h*)(p + 8);
  return u.v;
}

// B-fragment (32x16 KxN) from TRANSPOSED global storage BT[n][k], ld in halfs.
__device__ __forceinline__ v16h load_bt_frag(const _Float16* __restrict__ bt,
                                             int ld, int lane) {
  const int n = lane & 15, hl = lane >> 4;
  return *(const v16h*)(bt + (size_t)n * ld + hl * 16);
}

// B-fragment from the LDS-staged tile (row stride BROW_ halfs), column group j.
__device__ __forceinline__ v16h load_b_frag_lds(const _Float16* __restrict__ buf,
                                                int j, int lane) {
  const int nl = lane & 15, hl = lane >> 4;
  const _Float16* p = buf + (size_t)(j * 16 + nl) * BROW_ + hl * 16;
  union { v16h v; v8h h2[2]; } u;
  u.h2[0] = *(const v8h*)p;
  u.h2[1] = *(const v8h*)(p + 8);
  return u.v;
}

#ifdef HAVE_TDM
// Issue one TDM load: 2-D tile (32 halfs x 64 rows) from BT (row stride D_
// halfs) into LDS with 4-dword padding after every 16 dwords (-> BROW_ stride).
__device__ __forceinline__ void issue_tdm_b(const _Float16* __restrict__ gsrc,
                                            const _Float16* __restrict__ ldst,
                                            unsigned n_total) {
  unsigned long long ga = (unsigned long long)(uintptr_t)gsrc;
  unsigned lds_byte = (unsigned)(uintptr_t)ldst;   // flat LDS addr[31:0] = LDS offset
  v4u g0;
  g0[0] = 1u;                                       // count=1, user mode
  g0[1] = lds_byte;                                 // lds_addr
  g0[2] = (unsigned)ga;                             // global_addr[31:0]
  g0[3] = (unsigned)((ga >> 32) & 0x01FFFFFFu) | (2u << 30);  // addr[56:32] | type=2
  v8i g1;
  // data_size=2B(code1)<<16 | pad_enable<<20 | pad_interval(16dw,code3)<<22
  // | pad_amount(4dw,code3)<<25
  g1[0] = (int)0x06D10000u;
  g1[1] = (int)(((unsigned)D_ & 0xFFFFu) << 16);                 // tensor_dim0 lo16
  g1[2] = (int)((((unsigned)D_ >> 16) & 0xFFFFu) |
                ((n_total & 0xFFFFu) << 16));                    // dim0 hi | dim1 lo
  g1[3] = (int)(((n_total >> 16) & 0xFFFFu) | (32u << 16));      // dim1 hi | tile_dim0=32
  g1[4] = (int)64u;                                              // tile_dim1=64, tile_dim2=0
  g1[5] = (int)(unsigned)D_;                                     // tensor_dim0_stride lo32
  g1[6] = 0;                                                     // stride hi | dim1_stride lo
  g1[7] = 0;
  v4i z4 = {0, 0, 0, 0};
#if __clang_major__ >= 23
  v8i z8 = {0, 0, 0, 0, 0, 0, 0, 0};
  __builtin_amdgcn_tensor_load_to_lds(g0, g1, z4, z4, z8, 0);
#else
  __builtin_amdgcn_tensor_load_to_lds(g0, g1, z4, z4, 0);
#endif
}
#endif

// Shared GEMM mainloop: block = 4 waves, 64x64 tile; this wave's 16 rows start
// at Abase (ld = D_).  Bt0 = &BT[n0 * D_].  acc[4] = 16x64 accumulators.
__device__ __forceinline__ void gemm_mainloop(const _Float16* __restrict__ Abase,
                                              const _Float16* __restrict__ Bt0,
                                              unsigned n_total,
                                              _Float16* __restrict__ lds,
                                              v8f acc[4]) {
  const int tid = threadIdx.x;
  const int lane = tid & 31;
  const int NIT = D_ / 32;
#ifdef HAVE_TDM
  const bool w0 = (tid < 32);
  if (w0) issue_tdm_b(Bt0, lds, n_total);
  for (int i = 0; i < NIT; ++i) {
    const _Float16* cur = lds + (i & 1) * BBUF_;
    if (w0) {
      if (i + 1 < NIT) {
        issue_tdm_b(Bt0 + (i + 1) * 32, lds + ((i + 1) & 1) * BBUF_, n_total);
        __builtin_amdgcn_s_wait_tensorcnt(1);   // oldest (current buffer) done
      } else {
        __builtin_amdgcn_s_wait_tensorcnt(0);
      }
    }
    __syncthreads();                            // buffer ready for all waves
    v16h a = load_a_frag(Abase + i * 32, D_, lane);
#pragma unroll
    for (int j = 0; j < 4; ++j)
      acc[j] = wmma16(a, load_b_frag_lds(cur, j, lane), acc[j]);
    __syncthreads();                            // reads done before TDM reuses buffer
  }
#else
  (void)lds; (void)n_total;
  for (int i = 0; i < NIT; ++i) {
    v16h a = load_a_frag(Abase + i * 32, D_, lane);
#pragma unroll
    for (int j = 0; j < 4; ++j) {
      const _Float16* bt = Bt0 + (size_t)(j * 16) * D_ + i * 32;
      acc[j] = wmma16(a, load_bt_frag(bt, D_, lane), acc[j]);
    }
  }
#endif
}

// ---------------------------------------------------------------------------
// Packing / conversion kernels
// ---------------------------------------------------------------------------
__global__ void k_cvt_x(const float* __restrict__ x, _Float16* __restrict__ xh) {
  int i = blockIdx.x * blockDim.x + threadIdx.x;
  xh[i] = (_Float16)x[i];
}

__global__ void k_pack_w(const float* __restrict__ Wq, const float* __restrict__ Wk,
                         const float* __restrict__ Wv, _Float16* __restrict__ WT) {
  int i = blockIdx.x * blockDim.x + threadIdx.x;  // N3_*D_ threads
  int n = i >> 10, d = i & 1023;
  int sec = n >> 10, rem = n & 1023;
  int hh = rem >> 6, q = rem & 63;
  const float* src = (sec == 0) ? Wq : ((sec == 1) ? Wk : Wv);
  WT[i] = (_Float16)src[((size_t)hh * D_ + d) * QK_ + q];
}

__global__ void k_pack_bias(const float* __restrict__ bq, const float* __restrict__ bk,
                            const float* __restrict__ bv, float* __restrict__ bias) {
  int n = blockIdx.x * blockDim.x + threadIdx.x;
  if (n >= N3_) return;
  int sec = n >> 10, rem = n & 1023;
  int hh = rem >> 6, q = rem & 63;
  const float* src = (sec == 0) ? bq : ((sec == 1) ? bk : bv);
  bias[n] = src[hh * QK_ + q];
}

__global__ void k_pack_wff(const float* __restrict__ Wff, _Float16* __restrict__ WffT) {
  int i = blockIdx.x * blockDim.x + threadIdx.x;  // D_*D_ threads
  int n = i >> 10, d = i & 1023;
  WffT[i] = (_Float16)Wff[(size_t)d * D_ + n];
}

// ---------------------------------------------------------------------------
// QKV projection GEMM: 64x64 tile per block (4 waves), TDM-staged B.
// ---------------------------------------------------------------------------
__global__ void k_proj_gemm(const _Float16* __restrict__ xh,
                            const _Float16* __restrict__ WT,
                            const float* __restrict__ bias,
                            _Float16* __restrict__ Qh,   // [B,H,S,64]
                            _Float16* __restrict__ Kh,   // [B,H,S,64]
                            _Float16* __restrict__ VT) { // [B,H,64,S]
  __shared__ __align__(16) _Float16 btile[2 * BBUF_];
  const int tid = threadIdx.x, lane = tid & 31, wv = tid >> 5;
  const int mb = blockIdx.x % (BS_ / 64);
  const int nb = blockIdx.x / (BS_ / 64);        // 48 col-blocks
  const int m0 = mb * 64 + wv * 16, n0 = nb * 64;
  v8f acc[4] = {};
  gemm_mainloop(xh + (size_t)m0 * D_, WT + (size_t)n0 * D_, N3_, btile, acc);
  const int nl = lane & 15, hl = lane >> 4;
#pragma unroll
  for (int j = 0; j < 4; ++j) {
    int n = n0 + j * 16 + nl;
    int sec = n >> 10, rem = n & 1023;
    int hh = rem >> 6, q = rem & 63;
    float bi = bias[n];
#pragma unroll
    for (int r = 0; r < 8; ++r) {
      int m = m0 + r + 8 * hl;
      int bb = m >> 11, s = m & (S_ - 1);
      float v = acc[j][r] + bi;
      size_t bh = (size_t)bb * H_ + hh;
      if (sec == 0)      Qh[(bh * S_ + s) * QK_ + q] = (_Float16)v;
      else if (sec == 1) Kh[(bh * S_ + s) * QK_ + q] = (_Float16)v;
      else               VT[(bh * QK_ + q) * S_ + s] = (_Float16)v;
    }
  }
}

// ---------------------------------------------------------------------------
// Pass 1: per-COLUMN softmax stats (softmax axis = -2); column t valid for s<=t
// ---------------------------------------------------------------------------
__global__ void k_colstats(const _Float16* __restrict__ Qh,
                           const _Float16* __restrict__ Kh,
                           float* __restrict__ mcol, float* __restrict__ scol) {
  const int lane = threadIdx.x;
  const int tt = blockIdx.x % (S_ / 16);
  const int bh = blockIdx.x / (S_ / 16);
  const int t0 = tt * 16;
  const int nl = lane & 15, hl = lane >> 4;
  const int t = t0 + nl;
  const _Float16* kb = Kh + ((size_t)bh * S_ + t0) * QK_;
  v16h bk0 = load_bt_frag(kb, QK_, lane);
  v16h bk1 = load_bt_frag(kb + 32, QK_, lane);
  float m_run = -1e30f, l_run = 0.0f;
  for (int s0 = 0; s0 <= t0; s0 += 16) {
    const _Float16* qa = Qh + ((size_t)bh * S_ + s0) * QK_;
    v8f c = {};
    c = wmma16(load_a_frag(qa, QK_, lane), bk0, c);
    c = wmma16(load_a_frag(qa + 32, QK_, lane), bk1, c);
    float lm = -1e30f;
#pragma unroll
    for (int r = 0; r < 8; ++r) {
      int s = s0 + r + 8 * hl;
      if (s <= t) lm = fmaxf(lm, c[r]);
    }
    if (lm > -1e29f) {
      float nm = fmaxf(m_run, lm);
      l_run *= __expf(m_run - nm);
#pragma unroll
      for (int r = 0; r < 8; ++r) {
        int s = s0 + r + 8 * hl;
        if (s <= t) l_run += __expf(c[r] - nm);
      }
      m_run = nm;
    }
  }
  float m_o = __shfl_xor(m_run, 16, 32);
  float l_o = __shfl_xor(l_run, 16, 32);
  float m_f = fmaxf(m_run, m_o);
  float l_f = l_run * __expf(m_run - m_f) + l_o * __expf(m_o - m_f);
  if (lane < 16) {
    mcol[(size_t)bh * S_ + t] = m_f;
    scol[(size_t)bh * S_ + t] = 1.0f / l_f;
  }
}

// ---------------------------------------------------------------------------
// Pass 2: heads = P*V, P=exp(score-m[t])/l[t] (0 for s>t); P restaged via LDS
// ---------------------------------------------------------------------------
__global__ void k_attn(const _Float16* __restrict__ Qh,
                       const _Float16* __restrict__ Kh,
                       const _Float16* __restrict__ VT,
                       const float* __restrict__ mcol,
                       const float* __restrict__ scol,
                       float* __restrict__ combined) {  // [B,S,D]
  __shared__ __align__(32) _Float16 plds[16][40];
  const int lane = threadIdx.x;
  const int st = blockIdx.x % (S_ / 16);
  const int bh = blockIdx.x / (S_ / 16);
  const int b = bh / H_, hh = bh % H_;
  const int s0 = st * 16;
  const int nl = lane & 15, hl = lane >> 4;
  const _Float16* qa = Qh + ((size_t)bh * S_ + s0) * QK_;
  v16h a0 = load_a_frag(qa, QK_, lane);
  v16h a1 = load_a_frag(qa + 32, QK_, lane);
  const _Float16* vtb = VT + (size_t)bh * QK_ * S_;
  const float* mc = mcol + (size_t)bh * S_;
  const float* sc = scol + (size_t)bh * S_;
  v8f acc[4] = {};
  for (int t0 = s0 & ~31; t0 < S_; t0 += 32) {
    __syncthreads();
#pragma unroll
    for (int sub = 0; sub < 2; ++sub) {
      int tb = t0 + sub * 16;
      const _Float16* kb = Kh + ((size_t)bh * S_ + tb) * QK_;
      v8f c = {};
      c = wmma16(a0, load_bt_frag(kb, QK_, lane), c);
      c = wmma16(a1, load_bt_frag(kb + 32, QK_, lane), c);
      int t = tb + nl;
      float mt_ = mc[t], st_ = sc[t];
#pragma unroll
      for (int r = 0; r < 8; ++r) {
        int s = s0 + r + 8 * hl;
        float p = (s <= t) ? __expf(c[r] - mt_) * st_ : 0.0f;
        plds[r + 8 * hl][sub * 16 + nl] = (_Float16)p;
      }
    }
    __syncthreads();
    v16h ap;
    {
      union { v16h v; v8h h2[2]; } u;
      const _Float16* p = &plds[nl][hl * 16];
      u.h2[0] = *(const v8h*)p;
      u.h2[1] = *(const v8h*)(p + 8);
      ap = u.v;
    }
#pragma unroll
    for (int et = 0; et < 4; ++et) {
      v16h bv = load_bt_frag(vtb + (size_t)et * 16 * S_ + t0, S_, lane);
      acc[et] = wmma16(ap, bv, acc[et]);
    }
  }
#pragma unroll
  for (int et = 0; et < 4; ++et)
#pragma unroll
    for (int r = 0; r < 8; ++r) {
      int s = s0 + r + 8 * hl;
      combined[((size_t)b * S_ + s) * D_ + hh * QK_ + et * 16 + nl] = acc[et][r];
    }
}

// ---------------------------------------------------------------------------
// LayerNorm (unbiased std, ddof=1, eps added to std)
// ---------------------------------------------------------------------------
__global__ void k_ln1(const float* __restrict__ x, const float* __restrict__ comb,
                      float* __restrict__ y, _Float16* __restrict__ yh) {
  __shared__ float red[256];
  const int row = blockIdx.x, tid = threadIdx.x;
  const float* xr = x + (size_t)row * D_;
  const float* cr = comb + (size_t)row * D_;
  float vals[4]; float s = 0.0f;
#pragma unroll
  for (int i = 0; i < 4; ++i) { vals[i] = xr[tid + i * 256] + cr[tid + i * 256]; s += vals[i]; }
  red[tid] = s; __syncthreads();
  for (int o = 128; o > 0; o >>= 1) { if (tid < o) red[tid] += red[tid + o]; __syncthreads(); }
  float mean = red[0] / D_; __syncthreads();
  float sq = 0.0f;
#pragma unroll
  for (int i = 0; i < 4; ++i) { float d = vals[i] - mean; sq += d * d; }
  red[tid] = sq; __syncthreads();
  for (int o = 128; o > 0; o >>= 1) { if (tid < o) red[tid] += red[tid + o]; __syncthreads(); }
  float inv = 1.0f / (sqrtf(red[0] / (D_ - 1)) + 1e-4f);
#pragma unroll
  for (int i = 0; i < 4; ++i) {
    float o = (vals[i] - mean) * inv;
    y[(size_t)row * D_ + tid + i * 256] = o;
    yh[(size_t)row * D_ + tid + i * 256] = (_Float16)o;
  }
}

__global__ void k_ff_gemm(const _Float16* __restrict__ y1h,
                          const _Float16* __restrict__ WffT,
                          const float* __restrict__ bff,
                          float* __restrict__ ffout) {
  __shared__ __align__(16) _Float16 btile[2 * BBUF_];
  const int tid = threadIdx.x, lane = tid & 31, wv = tid >> 5;
  const int mb = blockIdx.x % (BS_ / 64);
  const int nb = blockIdx.x / (BS_ / 64);        // 16 col-blocks
  const int m0 = mb * 64 + wv * 16, n0 = nb * 64;
  v8f acc[4] = {};
  gemm_mainloop(y1h + (size_t)m0 * D_, WffT + (size_t)n0 * D_, D_, btile, acc);
  const int nl = lane & 15, hl = lane >> 4;
#pragma unroll
  for (int j = 0; j < 4; ++j) {
    int n = n0 + j * 16 + nl;
    float bi = bff[n];
#pragma unroll
    for (int r = 0; r < 8; ++r) {
      int m = m0 + r + 8 * hl;
      ffout[(size_t)m * D_ + n] = acc[j][r] + bi;
    }
  }
}

__global__ void k_ln2(const float* __restrict__ y1, const float* __restrict__ ff,
                      float* __restrict__ out) {
  __shared__ float red[256];
  const int row = blockIdx.x, tid = threadIdx.x;
  const float* ar = y1 + (size_t)row * D_;
  const float* br = ff + (size_t)row * D_;
  float vals[4]; float s = 0.0f;
#pragma unroll
  for (int i = 0; i < 4; ++i) { vals[i] = ar[tid + i * 256] + br[tid + i * 256]; s += vals[i]; }
  red[tid] = s; __syncthreads();
  for (int o = 128; o > 0; o >>= 1) { if (tid < o) red[tid] += red[tid + o]; __syncthreads(); }
  float mean = red[0] / D_; __syncthreads();
  float sq = 0.0f;
#pragma unroll
  for (int i = 0; i < 4; ++i) { float d = vals[i] - mean; sq += d * d; }
  red[tid] = sq; __syncthreads();
  for (int o = 128; o > 0; o >>= 1) { if (tid < o) red[tid] += red[tid + o]; __syncthreads(); }
  float inv = 1.0f / (sqrtf(red[0] / (D_ - 1)) + 1e-4f);
#pragma unroll
  for (int i = 0; i < 4; ++i)
    out[(size_t)row * D_ + tid + i * 256] = (vals[i] - mean) * inv;
}

// ---------------------------------------------------------------------------
extern "C" void kernel_launch(void* const* d_in, const int* in_sizes, int n_in,
                              void* d_out, int out_size, void* d_ws, size_t ws_size,
                              hipStream_t stream) {
  (void)in_sizes; (void)n_in; (void)out_size; (void)ws_size;
  const float* x   = (const float*)d_in[0];
  const float* Wq  = (const float*)d_in[1];
  const float* bq  = (const float*)d_in[2];
  const float* Wk  = (const float*)d_in[3];
  const float* bk  = (const float*)d_in[4];
  const float* Wv  = (const float*)d_in[5];
  const float* bv  = (const float*)d_in[6];
  const float* Wff = (const float*)d_in[7];
  const float* bff = (const float*)d_in[8];
  float* out = (float*)d_out;

  char* w = (char*)d_ws;
  size_t off = 0;
  auto grab = [&](size_t bytes) { char* p = w + off; off += (bytes + 255) & ~(size_t)255; return p; };
  _Float16* xh    = (_Float16*)grab((size_t)BS_ * D_ * 2);
  _Float16* WT    = (_Float16*)grab((size_t)N3_ * D_ * 2);
  float*    biasA = (float*)   grab((size_t)N3_ * 4);
  _Float16* WffT  = (_Float16*)grab((size_t)D_ * D_ * 2);
  _Float16* Qh    = (_Float16*)grab((size_t)B_ * H_ * S_ * QK_ * 2);
  _Float16* Kh    = (_Float16*)grab((size_t)B_ * H_ * S_ * QK_ * 2);
  _Float16* VT    = (_Float16*)grab((size_t)B_ * H_ * QK_ * S_ * 2);
  float*    mcol  = (float*)   grab((size_t)B_ * H_ * S_ * 4);
  float*    scol  = (float*)   grab((size_t)B_ * H_ * S_ * 4);
  float*    comb  = (float*)   grab((size_t)BS_ * D_ * 4);
  float*    y1    = (float*)   grab((size_t)BS_ * D_ * 4);
  _Float16* y1h   = (_Float16*)grab((size_t)BS_ * D_ * 2);
  float*    ffout = (float*)   grab((size_t)BS_ * D_ * 4);

  // pack / convert
  k_cvt_x    <<<(BS_ * D_) / 256, 256, 0, stream>>>(x, xh);
  k_pack_w   <<<(N3_ * D_) / 256, 256, 0, stream>>>(Wq, Wk, Wv, WT);
  k_pack_bias<<<(N3_ + 255) / 256, 256, 0, stream>>>(bq, bk, bv, biasA);
  k_pack_wff <<<(D_ * D_) / 256, 256, 0, stream>>>(Wff, WffT);

  // QKV projections: 64x64 tiles, 4 waves/block, TDM-staged B
  k_proj_gemm<<<(BS_ / 64) * (N3_ / 64), 128, 0, stream>>>(xh, WT, biasA, Qh, Kh, VT);

  // attention: column softmax stats then P*V (both WMMA)
  const int attn_blocks = B_ * H_ * (S_ / 16);
  k_colstats<<<attn_blocks, 32, 0, stream>>>(Qh, Kh, mcol, scol);
  k_attn    <<<attn_blocks, 32, 0, stream>>>(Qh, Kh, VT, mcol, scol, comb);

  // residual + LN1
  k_ln1<<<BS_, 256, 0, stream>>>(x, comb, y1, y1h);

  // feed-forward GEMM
  k_ff_gemm<<<(BS_ / 64) * (D_ / 64), 128, 0, stream>>>(y1h, WffT, bff, ffout);

  // residual + LN2 -> output
  k_ln2<<<BS_, 256, 0, stream>>>(y1, ffout, out);
}
